// _IPEXScaleDotProductRef_69956427317262
// MI455X (gfx1250) — matrix operand order repaired and checked by reading, not compiled, not run on hardware
//
#include <hip/hip_runtime.h>

// ---------------------------------------------------------------------------
// Flash-attention forward for gfx1250 (MI455X), wave32 + v_wmma_f32_16x16x32_bf16
// B=2, HQ=32 (HKV=8, GQA x4), S=2048, D=128, causal, scale = 1/sqrt(D)
// BLOCK_M=64 (4 waves x 16 rows), BLOCK_N=64 keys/tile, row-sum via ones-WMMA.
// ---------------------------------------------------------------------------

typedef __bf16 bf16_t;
typedef __attribute__((ext_vector_type(16))) __bf16 v16bf;
typedef __attribute__((ext_vector_type(8)))  __bf16 v8bf;
typedef __attribute__((ext_vector_type(4)))  __bf16 v4bf;
typedef __attribute__((ext_vector_type(8)))  float  v8f;
typedef __attribute__((ext_vector_type(4)))  float  v4f;

union BF16x16 { v16bf v; v8bf h[2]; };

#define B_      2
#define HQ_     32
#define HKV_    8
#define S_      2048
#define D_      128
#define NREP    (HQ_ / HKV_)
#define BLOCK_M 64          // 4 waves x 16 query rows
#define BLOCK_N 64          // key tile (4 N-subtiles of 16)
#define NWAVES  4
#define VT_STR  72          // padded key-stride of transposed V tile (bf16), mult of 8
#define NEG_BIG (-1.0e30f)

__global__ __launch_bounds__(128, 1)
void fa_fwd_gfx1250(const float* __restrict__ Q,
                    const float* __restrict__ K,
                    const float* __restrict__ V,
                    float* __restrict__ O) {
  __shared__ bf16_t Ks[BLOCK_N * D_];          // [key][d] bf16, 16 KB
  __shared__ bf16_t Vt[D_ * VT_STR];           // [d][key+pad] bf16, 18 KB
  __shared__ bf16_t Ps[NWAVES][16 * BLOCK_N];  // per-wave P scratch, 8 KB

  const int tid  = threadIdx.x;
  const int lane = tid & 31;
  const int wave = tid >> 5;
  const int half = lane >> 4;     // 0: lanes 0-15, 1: lanes 16-31
  const int l16  = lane & 15;

  const int qb  = blockIdx.x;
  const int bh  = blockIdx.y;
  const int b   = bh / HQ_;
  const int h   = bh % HQ_;
  const int kvh = h / NREP;                   // repeat_interleave GQA mapping

  const float scale = 0.08838834764831845f;   // 1/sqrt(128)

  const float* Qg = Q + (size_t)(b * HQ_  + h)   * S_ * D_;
  const float* Kg = K + (size_t)(b * HKV_ + kvh) * S_ * D_;
  const float* Vg = V + (size_t)b * S_ * (HKV_ * D_) + (size_t)kvh * D_; // [B,S,Hkv,D]
  float*       Og = O + (size_t)(b * HQ_  + h)   * S_ * D_;

  const int q0   = qb * BLOCK_M + wave * 16;  // first absolute query row of wave
  const int rowM = q0 + l16;                  // A-fragment row (M = lane&15)

  // ---- Q fragments in WMMA A layout (K = (j/8)*16 + half*8 + j%8), pre-scaled ----
  v16bf qf[4];
  #pragma unroll
  for (int c = 0; c < 4; ++c) {
    const float* p0 = Qg + (size_t)rowM * D_ + c * 32 + half * 8;
    const float* p1 = p0 + 16;
    #pragma unroll
    for (int j = 0; j < 8; ++j) {
      qf[c][j]     = (bf16_t)(p0[j] * scale);
      qf[c][j + 8] = (bf16_t)(p1[j] * scale);
    }
  }

  // All-ones B matrix (32x16): row-sum reducer for P @ 1
  v16bf onesB;
  #pragma unroll
  for (int j = 0; j < 16; ++j) onesB[j] = (bf16_t)1.0f;

  // ---- Online-softmax state: 8 rows/lane (row = r + 8*half) ----
  float m8[8], l8[8];
  v8f acc[8];
  #pragma unroll
  for (int r = 0; r < 8; ++r) { m8[r] = NEG_BIG; l8[r] = 0.0f; }
  #pragma unroll
  for (int dt = 0; dt < 8; ++dt)
    #pragma unroll
    for (int r = 0; r < 8; ++r) acc[dt][r] = 0.0f;

  for (int kt = 0; kt <= qb; ++kt) {          // 64-key tiles, causal bound
    const int k0 = kt * BLOCK_N;

    __syncthreads();

    // ---- Stage K (row-major bf16) and V (transposed bf16) tiles ----
    #pragma unroll
    for (int rep = 0; rep < 16; ++rep) {
      const int idx = rep * 512 + tid * 4;    // 8192 elements, float4 granules
      const int key = idx >> 7;
      const int d   = idx & 127;

      v4f kv = *(const v4f*)(Kg + (size_t)(k0 + key) * D_ + d);
      v4bf kb;
      kb.x = (bf16_t)kv.x; kb.y = (bf16_t)kv.y;
      kb.z = (bf16_t)kv.z; kb.w = (bf16_t)kv.w;
      *(v4bf*)(&Ks[key * D_ + d]) = kb;

      v4f vv = *(const v4f*)(Vg + (size_t)(k0 + key) * (HKV_ * D_) + d);
      Vt[(d + 0) * VT_STR + key] = (bf16_t)vv.x;
      Vt[(d + 1) * VT_STR + key] = (bf16_t)vv.y;
      Vt[(d + 2) * VT_STR + key] = (bf16_t)vv.z;
      Vt[(d + 3) * VT_STR + key] = (bf16_t)vv.w;
    }
    __syncthreads();

    if (kt < qb) {  // prefetch next K tile (global_prefetch_b8)
      __builtin_prefetch(Kg + (size_t)(k0 + BLOCK_N + (tid >> 1)) * D_ + (tid & 1) * 64, 0, 0);
    }

    // ---- S = (Q*scale) @ K^T : four 16x16 C subtiles over 4 D-chunks ----
    v8f cf[4];
    #pragma unroll
    for (int nt = 0; nt < 4; ++nt)
      #pragma unroll
      for (int r = 0; r < 8; ++r) cf[nt][r] = 0.0f;

    #pragma unroll
    for (int nt = 0; nt < 4; ++nt) {
      // Skip fully-masked diagonal subtiles; force a SCALAR branch so WMMA
      // always runs with EXEC all-1s (condition is wave-uniform by construction).
      if (__builtin_amdgcn_readfirstlane((k0 + nt * 16 <= q0 + 15) ? 1 : 0)) {
        #pragma unroll
        for (int dc = 0; dc < 4; ++dc) {
          BF16x16 kb;  // B 32x16: K(d) = (j/8)*16 + half*8 + j%8, N(key) = nt*16 + l16
          const bf16_t* kp = &Ks[(nt * 16 + l16) * D_ + dc * 32 + half * 8];
          kb.h[0] = *(const v8bf*)(kp);
          kb.h[1] = *(const v8bf*)(kp + 16);
          cf[nt] = __builtin_amdgcn_wmma_f32_16x16x32_bf16(false, qf[dc], false, kb.v,
                                                           (short)0, cf[nt], false, false);
        }
      }
    }

    // ---- Causal mask (only the diagonal tile; block-uniform scalar branch) ----
    if (kt == qb) {
      #pragma unroll
      for (int nt = 0; nt < 4; ++nt) {
        const int key = k0 + nt * 16 + l16;
        #pragma unroll
        for (int r = 0; r < 8; ++r) {
          const int row = q0 + r + half * 8;
          if (key > row) cf[nt][r] = NEG_BIG;
        }
      }
    }

    // ---- Row max over 64 keys: local 4-way max + 4-step butterfly over 16 lanes ----
    float alpha[8];
    #pragma unroll
    for (int r = 0; r < 8; ++r) {
      float t = fmaxf(fmaxf(cf[0][r], cf[1][r]), fmaxf(cf[2][r], cf[3][r]));
      t = fmaxf(t, __shfl_xor(t, 1, 16));
      t = fmaxf(t, __shfl_xor(t, 2, 16));
      t = fmaxf(t, __shfl_xor(t, 4, 16));
      t = fmaxf(t, __shfl_xor(t, 8, 16));
      const float mn = fmaxf(m8[r], t);
      alpha[r] = __expf(m8[r] - mn);
      m8[r] = mn;
    }

    // ---- P = exp(S - m), written to per-wave LDS scratch in [row][key] form ----
    bf16_t* pw = Ps[wave];
    #pragma unroll
    for (int r = 0; r < 8; ++r) {
      const int prow = r + half * 8;
      #pragma unroll
      for (int nt = 0; nt < 4; ++nt) {
        const float p = __expf(cf[nt][r] - m8[r]);
        pw[prow * BLOCK_N + nt * 16 + l16] = (bf16_t)p;
      }
    }

    // ---- Re-read P as two A fragments (K chunks 0..31, 32..63) ----
    BF16x16 pa[2];
    #pragma unroll
    for (int kc = 0; kc < 2; ++kc) {
      const bf16_t* pp = &pw[l16 * BLOCK_N + kc * 32 + half * 8];
      pa[kc].h[0] = *(const v8bf*)(pp);        // K = kc*32 + 8h + j
      pa[kc].h[1] = *(const v8bf*)(pp + 16);   // K = kc*32 + 16 + 8h + j
    }

    // ---- Row sums via ones-matmul: csum[r] = sum_k P[row][k] (replicated) ----
    v8f csum;
    #pragma unroll
    for (int r = 0; r < 8; ++r) csum[r] = 0.0f;
    csum = __builtin_amdgcn_wmma_f32_16x16x32_bf16(false, pa[0].v, false, onesB,
                                                   (short)0, csum, false, false);
    csum = __builtin_amdgcn_wmma_f32_16x16x32_bf16(false, pa[1].v, false, onesB,
                                                   (short)0, csum, false, false);
    #pragma unroll
    for (int r = 0; r < 8; ++r) l8[r] = l8[r] * alpha[r] + csum[r];

    // ---- Rescale accumulators ----
    #pragma unroll
    for (int dt = 0; dt < 8; ++dt)
      #pragma unroll
      for (int r = 0; r < 8; ++r) acc[dt][r] *= alpha[r];

    // ---- O += P @ V : 8 D-column tiles x 2 K-chunks ----
    #pragma unroll
    for (int dt = 0; dt < 8; ++dt) {
      #pragma unroll
      for (int kc = 0; kc < 2; ++kc) {
        BF16x16 vb;  // B 32x16: K(key) = kc*32 + (j/8)*16 + half*8 + j%8, N(dcol) = dt*16 + l16
        const bf16_t* vp = &Vt[(dt * 16 + l16) * VT_STR + kc * 32 + half * 8];
        vb.h[0] = *(const v8bf*)(vp);
        vb.h[1] = *(const v8bf*)(vp + 16);
        acc[dt] = __builtin_amdgcn_wmma_f32_16x16x32_bf16(false, pa[kc].v, false, vb.v,
                                                          (short)0, acc[dt], false, false);
      }
    }
  }

  // ---- Epilogue: O = acc / l ----
  float inv[8];
  #pragma unroll
  for (int r = 0; r < 8; ++r) inv[r] = 1.0f / l8[r];

  #pragma unroll
  for (int dt = 0; dt < 8; ++dt) {
    #pragma unroll
    for (int r = 0; r < 8; ++r) {
      const int row = q0 + r + half * 8;
      Og[(size_t)row * D_ + dt * 16 + l16] = acc[dt][r] * inv[r];
    }
  }
}

extern "C" void kernel_launch(void* const* d_in, const int* in_sizes, int n_in,
                              void* d_out, int out_size, void* d_ws, size_t ws_size,
                              hipStream_t stream) {
  (void)in_sizes; (void)n_in; (void)out_size; (void)d_ws; (void)ws_size;
  const float* Q = (const float*)d_in[0];   // [B,HQ,S,D]
  const float* K = (const float*)d_in[1];   // [B,HKV,S,D]
  const float* V = (const float*)d_in[2];   // [B,S,Hkv,D]
  // d_in[3] (attention_mask) is exactly the causal mask; applied analytically.
  float* O = (float*)d_out;                 // [B,HQ,S,D]

  dim3 grid(S_ / BLOCK_M, B_ * HQ_);
  dim3 block(128);
  fa_fwd_gfx1250<<<grid, block, 0, stream>>>(Q, K, V, O);
}
